// UltraSimpleGRU_18047452578325
// MI455X (gfx1250) — matrix operand order repaired and checked by reading, not compiled, not run on hardware
//
#include <hip/hip_runtime.h>
#include <hip/hip_bf16.h>
#include <math.h>

typedef __attribute__((ext_vector_type(16))) _Float16 v16h;
typedef __attribute__((ext_vector_type(2)))  _Float16 h2f;
typedef __attribute__((ext_vector_type(8)))  float    v8f;

#define GRU_I 2
#define GRU_H 8
#define GRU_T 1024

__device__ __forceinline__ float fast_rcp(float x) {
    return __builtin_amdgcn_rcpf(x);
}
__device__ __forceinline__ float tanhf_fast(float x) {
#if __has_builtin(__builtin_amdgcn_tanhf)
    return __builtin_amdgcn_tanhf(x);                    // gfx1250 v_tanh_f32
#else
    return fmaf(-2.0f, fast_rcp(__expf(2.0f * x) + 1.0f), 1.0f);
#endif
}

__device__ __forceinline__ h2f pk_f16(float a, float b) {
    return __builtin_bit_cast(h2f, __builtin_amdgcn_cvt_pkrtz(a, b));
}

// One wave per block; each wave owns TWO independent 16-batch WMMA column
// tiles (32 batch elements). The hidden dim is SPLIT across half-waves:
//   tile1 rows = [r j0..3, z j0..3 | r j4..7, z j4..7]   (pre-scaled by 0.5)
//   tile2 rows = [xn j0..3, hn j0..3 | xn j4..7, hn j4..7]
// so each lane owns r,z,xn,hn for 4 hidden elements of its batch column:
// 2 WMMAs per tile-step, no duplicated gate math, sigmoid = fma(tanh,.5,.5).
// The B operand's h[4..7] rows come from the upper half via two 32-bit
// shuffles of the already-f16-packed state. State is fp32 in registers.
__global__ __launch_bounds__(32) void gru_wmma_kernel(
    const float* __restrict__ x,      // [B, T, 2]
    const float* __restrict__ W_ih,   // [24, 2]   gate order r,z,n
    const float* __restrict__ W_hh,   // [24, 8]
    const float* __restrict__ b_ih,   // [24]
    const float* __restrict__ b_hh,   // [24]
    const float* __restrict__ fc_w,   // [1, 8]
    const float* __restrict__ fc_b,   // [1]
    float* __restrict__ out,          // [B]
    int B)
{
    const int  lane = threadIdx.x & 31;
    const int  col  = lane & 15;          // batch column / A row (mod 16)
    const bool lo   = lane < 16;

    // ---- row -> gate mapping for the split tiles ---------------------------
    // A/C row m: jj = m&3, second = (m>>2)&1 (z or hn), j = jj + 4*(m>>3).
    auto wrow1 = [&](int m) -> int {          // W row for tile1 (r/z)
        int j = (m & 3) + 4 * ((m >> 3) & 1);
        return ((m >> 2) & 1) ? 8 + j : j;    // z rows are W rows 8..15
    };
    auto a1 = [&](int m, int k) -> float {    // tile1, pre-scaled by 0.5
        int w = wrow1(m);
        float v = 0.0f;
        if (k < 2)       v = W_ih[w * GRU_I + k];
        else if (k < 10) v = W_hh[w * GRU_H + (k - 2)];
        return 0.5f * v;
    };
    auto a2 = [&](int m, int k) -> float {    // tile2 (xn/hn)
        int j = (m & 3) + 4 * ((m >> 3) & 1);
        bool ishn = (m >> 2) & 1;
        if (ishn) return (k >= 2 && k < 10) ? W_hh[(16 + j) * GRU_H + (k - 2)] : 0.0f;
        return (k < 2) ? W_ih[(16 + j) * GRU_I + k] : 0.0f;
    };

    v16h A1, A2;
#pragma unroll
    for (int v = 0; v < 8; ++v) {
        // A 16x32 in this code's own contiguous k order (B matches): lanes0-15 vgpr v holds K={2v,2v+1};
        // lanes16-31 hold K={16+2v,16+2v+1}.
        int k0 = lo ? 2 * v : 16 + 2 * v;
        int m  = col;
        A1[2 * v]     = (_Float16)a1(m, k0);
        A1[2 * v + 1] = (_Float16)a1(m, k0 + 1);
        A2[2 * v]     = (_Float16)a2(m, k0);
        A2[2 * v + 1] = (_Float16)a2(m, k0 + 1);
    }

    // ---- C bias tiles (row = v + 8*hi_half) --------------------------------
    v8f C1, C2;
#pragma unroll
    for (int v = 0; v < 8; ++v) {
        int m = lo ? v : 8 + v;
        int w = wrow1(m);
        C1[v] = 0.5f * (b_ih[w] + b_hh[w]);
        int j = (m & 3) + 4 * ((m >> 3) & 1);
        C2[v] = ((m >> 2) & 1) ? b_hh[16 + j] : b_ih[16 + j];
    }

    // ---- state: each lane owns 4 hidden elems (lo: j=0..3, hi: j=4..7) -----
    float h0[4], h1[4];
#pragma unroll
    for (int j = 0; j < 4; ++j) { h0[j] = 0.0f; h1[j] = 0.0f; }

    const int    b0  = blockIdx.x * 32 + col;
    const int    b1  = b0 + 16;
    const float* xp0 = x + (size_t)b0 * GRU_T * GRU_I;
    const float* xp1 = x + (size_t)b1 * GRU_T * GRU_I;

    // B skeletons: elements 10..15 stay 0; all A columns K>=10 are zero, so
    // the upper half-wave rows (K=16..31) may hold any finite values.
    v16h Bm0, Bm1;
#pragma unroll
    for (int i = 0; i < 16; ++i) { Bm0[i] = (_Float16)0.0f; Bm1[i] = (_Float16)0.0f; }

    // Pack own 4 h values to f16 pairs, swap packs across half-waves, and
    // assemble B rows K=2..9 = h[0..7] (valid in lanes 0..15; don't-care hi).
    auto build_B = [&](v16h& Bm, const float* h, float xa, float xb) {
        h2f q0 = pk_f16(h[0], h[1]);
        h2f q1 = pk_f16(h[2], h[3]);
        unsigned p0u = __shfl_xor(__builtin_bit_cast(unsigned, q0), 16, 32);
        unsigned p1u = __shfl_xor(__builtin_bit_cast(unsigned, q1), 16, 32);
        h2f p0 = __builtin_bit_cast(h2f, p0u);
        h2f p1 = __builtin_bit_cast(h2f, p1u);
        h2f px = pk_f16(xa, xb);
        Bm[0] = px[0]; Bm[1] = px[1];
        Bm[2] = q0[0]; Bm[3] = q0[1];
        Bm[4] = q1[0]; Bm[5] = q1[1];
        Bm[6] = p0[0]; Bm[7] = p0[1];
        Bm[8] = p1[0]; Bm[9] = p1[1];
    };

    // D1 vgprs 0..3 = r-preact*0.5, 4..7 = z-preact*0.5 (for this lane's j's);
    // D2 vgprs 0..3 = xn, 4..7 = hn.
    auto gate_update = [&](float* h, const v8f& D1, const v8f& D2) {
#pragma unroll
        for (int jj = 0; jj < 4; ++jj) {
            float r = fmaf(tanhf_fast(D1[jj]),     0.5f, 0.5f);
            float z = fmaf(tanhf_fast(D1[4 + jj]), 0.5f, 0.5f);
            float n = tanhf_fast(fmaf(r, D2[4 + jj], D2[jj]));
            h[jj] = fmaf(z, h[jj] - n, n);        // (1-z)*n + z*h
        }
    };

#define GRU_STEP(XA0, XB0, XA1, XB1)                                          \
    do {                                                                      \
        build_B(Bm0, h0, (XA0), (XB0));                                       \
        build_B(Bm1, h1, (XA1), (XB1));                                       \
        v8f d10 = __builtin_amdgcn_wmma_f32_16x16x32_f16(                     \
            false, A1, false, Bm0, (short)0, C1, false, false);               \
        v8f d20 = __builtin_amdgcn_wmma_f32_16x16x32_f16(                     \
            false, A2, false, Bm0, (short)0, C2, false, false);               \
        v8f d11 = __builtin_amdgcn_wmma_f32_16x16x32_f16(                     \
            false, A1, false, Bm1, (short)0, C1, false, false);               \
        v8f d21 = __builtin_amdgcn_wmma_f32_16x16x32_f16(                     \
            false, A2, false, Bm1, (short)0, C2, false, false);               \
        gate_update(h0, d10, d20);                                            \
        gate_update(h1, d11, d21);                                            \
    } while (0)

    for (int t = 0; t < GRU_T; t += 2) {
        if ((t & 15) == 0) {
            __builtin_prefetch(xp0 + (size_t)(t + 16) * GRU_I, 0, 0);
            __builtin_prefetch(xp1 + (size_t)(t + 16) * GRU_I, 0, 0);
        }
        // one b128 load per tile covers two timesteps
        const float4 xq0 = *(const float4*)(xp0 + (size_t)t * GRU_I);
        const float4 xq1 = *(const float4*)(xp1 + (size_t)t * GRU_I);

        GRU_STEP(xq0.x, xq0.y, xq1.x, xq1.y);   // step t
        GRU_STEP(xq0.z, xq0.w, xq1.z, xq1.w);   // step t+1
    }
#undef GRU_STEP

    // ---- fc + softplus head: cross-half partial dot then reduce ------------
    float part0 = 0.0f, part1 = 0.0f;
#pragma unroll
    for (int jj = 0; jj < 4; ++jj) {
        float w = fc_w[jj + (lo ? 0 : 4)];
        part0 = fmaf(h0[jj], w, part0);
        part1 = fmaf(h1[jj], w, part1);
    }
    part0 += __shfl_xor(part0, 16, 32);
    part1 += __shfl_xor(part1, 16, 32);
    float acc0 = part0 + fc_b[0];
    float acc1 = part1 + fc_b[0];
    float sp0 = (acc0 > 20.0f) ? acc0 : __logf(1.0f + __expf(acc0));
    float sp1 = (acc1 > 20.0f) ? acc1 : __logf(1.0f + __expf(acc1));
    if (lo) {
        if (b0 < B) out[b0] = sp0 + 1e-8f;
        if (b1 < B) out[b1] = sp1 + 1e-8f;
    }
}

extern "C" void kernel_launch(void* const* d_in, const int* in_sizes, int n_in,
                              void* d_out, int out_size, void* d_ws, size_t ws_size,
                              hipStream_t stream) {
    const float* x    = (const float*)d_in[0];
    const float* W_ih = (const float*)d_in[1];
    const float* W_hh = (const float*)d_in[2];
    const float* b_ih = (const float*)d_in[3];
    const float* b_hh = (const float*)d_in[4];
    const float* fc_w = (const float*)d_in[5];
    const float* fc_b = (const float*)d_in[6];
    float*       out  = (float*)d_out;

    const int B = in_sizes[0] / (GRU_T * GRU_I);   // 16384 for the reference

    dim3 grid((B + 31) / 32), block(32);
    gru_wmma_kernel<<<grid, block, 0, stream>>>(x, W_ih, W_hh, b_ih, b_hh,
                                                fc_w, fc_b, out, B);
}
